// VanillaLSTM_63453846831597
// MI455X (gfx1250) — matrix operand order repaired and checked by reading, not compiled, not run on hardware
//
#include <hip/hip_runtime.h>

// Problem constants (match reference)
#define B_   64
#define T_   512
#define D_   1024
#define H_   2048
#define G4H_ (4 * H_)

#define KC_X   (D_ / 32)            // 32 K-chunks over the x projection
#define KC_H   (H_ / 32)            // 64 K-chunks over the recurrence
#define KC_ALL (KC_X + KC_H)        // 96 total

// ---- CDNA5 WMMA types -------------------------------------------------------
typedef __bf16 v16bf __attribute__((ext_vector_type(16)));
typedef __bf16 v8bf  __attribute__((ext_vector_type(8)));
typedef __bf16 v4bf  __attribute__((ext_vector_type(4)));
typedef float  v8f   __attribute__((ext_vector_type(8)));

static __device__ __forceinline__ v8f wmma_bf16(v16bf a, v16bf b, v8f c) {
    // 8 args: (neg_a, A, neg_b, B, c_mod, C, reuse_a, reuse_b)
    return __builtin_amdgcn_wmma_f32_16x16x32_bf16(false, a, false, b,
                                                   (short)0, c, false, false);
}

// A-fragment: 16x32 (MxK), row-major source with row stride `ld` (elements).
// Contiguous K order (B loads match it): lane L = row (L&15); lanes<16 hold K=0..15; lanes>=16
// hold K=16..31 -> two contiguous 16B chunks per lane.
static __device__ __forceinline__ v16bf load_frag_A(const __bf16* __restrict__ base,
                                                    size_t ld, int m0, int k0, int lane) {
    int m  = m0 + (lane & 15);
    int hi = lane >> 4;
    const __bf16* p = base + (size_t)m * ld + (size_t)(k0 + hi * 16);
    v8bf lo = *(const v8bf*)p;
    v8bf hh = *(const v8bf*)(p + 8);
    return __builtin_shufflevector(lo, hh, 0, 1, 2, 3, 4, 5, 6, 7,
                                           8, 9, 10, 11, 12, 13, 14, 15);
}

static __device__ __forceinline__ float fsigmoid(float x) {
    return 1.0f / (1.0f + __expf(-x));
}
static __device__ __forceinline__ float ftanh(float x) {
    float e = __expf(-2.0f * fabsf(x));
    float t = (1.0f - e) / (1.0f + e);
    return copysignf(t, x);
}

// ---- setup kernels ----------------------------------------------------------
__global__ void k_cvt_bf16(const float* __restrict__ in, __bf16* __restrict__ out, long n) {
    long i = ((long)blockIdx.x * blockDim.x + threadIdx.x) * 4;
    if (i + 3 < n) {
        float4 v = *(const float4*)(in + i);
        v4bf o;
        o.x = (__bf16)v.x; o.y = (__bf16)v.y; o.z = (__bf16)v.z; o.w = (__bf16)v.w;
        *(v4bf*)(out + i) = o;
    } else {
        for (long j = i; j < n; ++j) out[j] = (__bf16)in[j];
    }
}

__global__ void k_bias_sum(const float* __restrict__ b_ih, const float* __restrict__ b_hh,
                           float* __restrict__ bias, int n) {
    int i = blockIdx.x * blockDim.x + threadIdx.x;
    if (i < n) bias[i] = b_ih[i] + b_hh[i];
}

__global__ void k_zero_u32(unsigned int* __restrict__ p, long n) {
    long i = (long)blockIdx.x * blockDim.x + threadIdx.x;
    if (i < n) p[i] = 0u;
}

// ---- one LSTM time step -----------------------------------------------------
// grid = H/16 blocks (n-tiles), block = 128 threads = 4 waves (m-tiles of B=64).
// All 4 waves share the same weight columns n0, so the 4-gate weight tile
// (4 gates x 16 rows x 32 K = 4 KB bf16 per chunk) is staged once per block
// into LDS (double-buffered), instead of each wave fetching it redundantly.
// Global loads (A-fragment + next-chunk B prefetch) are issued BEFORE the
// barrier each iteration — they don't touch LDS, so they overlap the barrier
// and the WMMA consumption of the current LDS buffer.
__global__ void __launch_bounds__(128)
k_lstm_step(const __bf16* __restrict__ x_bf,    // [B, T, D]
            const __bf16* __restrict__ Wih,     // [4H, D]
            const __bf16* __restrict__ Whh,     // [4H, H]
            const float*  __restrict__ bias,    // [4H]
            const __bf16* __restrict__ h_in,    // [B, H]
            const float*  __restrict__ c_in,    // [B, H]
            __bf16* __restrict__ h_out,         // [B, H]
            float*  __restrict__ c_out,         // [B, H]
            int t) {
    const int tid  = threadIdx.x;
    const int lane = tid & 31;
    const int wave = tid >> 5;
    const int m0 = wave * 16;          // batch tile
    const int n0 = blockIdx.x * 16;    // hidden-column tile

    // LDS: double-buffered 4-gate B tile. [buf][g*512 + n*32 + k], 8 KB total.
    __shared__ __bf16 Bs[2][4 * 16 * 32];

    // staging address for this thread's 16-element (32 B) chunk of chunk `kc`
    const int sg    = tid >> 5;          // gate
    const int sn    = (tid & 31) >> 1;   // row within tile
    const int shalf = tid & 1;           // low/high 16 K elements
    const __bf16* sp_x = Wih + ((size_t)sg * H_ + (size_t)(n0 + sn)) * D_
                             + (size_t)(shalf * 16);
    const __bf16* sp_h = Whh + ((size_t)sg * H_ + (size_t)(n0 + sn)) * H_
                             + (size_t)(shalf * 16);
    auto stage_ptr = [&](int kc) -> const __bf16* {
        return (kc < KC_X) ? (sp_x + (size_t)kc * 32)
                           : (sp_h + (size_t)(kc - KC_X) * 32);
    };
    const __bf16* xt = x_bf + (size_t)t * D_;
    auto a_frag = [&](int kc) -> v16bf {
        if (kc < KC_X) return load_frag_A(xt, (size_t)T_ * D_, m0, kc * 32, lane);
        return load_frag_A(h_in, H_, m0, (kc - KC_X) * 32, lane);
    };

    v8f acc[4] = {v8f{}, v8f{}, v8f{}, v8f{}};

    // prologue: stage chunk 0 into buffer 0
    {
        v16bf r0 = *(const v16bf*)stage_ptr(0);
        *(v16bf*)(&Bs[0][tid * 16]) = r0;
    }

    const int boff = (lane & 15) * 32 + (lane >> 4) * 16;
#pragma unroll 1
    for (int kc = 0; kc < KC_ALL; ++kc) {
        const int cur = kc & 1;

        // global loads first: they don't depend on LDS state
        v16bf breg{};
        const bool have_next = (kc + 1) < KC_ALL;
        if (have_next) breg = *(const v16bf*)stage_ptr(kc + 1);
        v16bf a = a_frag(kc);

        __syncthreads();  // LDS[cur] ready; LDS[cur^1] fully consumed

        // consume the staged B tile from LDS: 4 gates, accumulate in place
#pragma unroll
        for (int g = 0; g < 4; ++g) {
            v16bf b = *(const v16bf*)(&Bs[cur][g * 512 + boff]);
            acc[g] = wmma_bf16(a, b, acc[g]);
        }

        if (have_next) *(v16bf*)(&Bs[cur ^ 1][tid * 16]) = breg;
    }

    // elementwise LSTM cell update (fp32). C/D layout: lanes 0-15: N=lane,
    // VGPR r -> M=r; lanes 16-31: N=lane-16, VGPR r -> M=8+r.
    const int hi = lane >> 4;
    const int n  = n0 + (lane & 15);
    const float bi = bias[0 * H_ + n];
    const float bf = bias[1 * H_ + n];
    const float bg = bias[2 * H_ + n];
    const float bo = bias[3 * H_ + n];
#pragma unroll
    for (int r = 0; r < 8; ++r) {
        const int m = m0 + r + 8 * hi;
        const size_t idx = (size_t)m * H_ + n;
        float iv = fsigmoid(acc[0][r] + bi);
        float fv = fsigmoid(acc[1][r] + bf);
        float gv = ftanh(acc[2][r] + bg);
        float ov = fsigmoid(acc[3][r] + bo);
        float cv = fv * c_in[idx] + iv * gv;
        c_out[idx] = cv;
        h_out[idx] = (__bf16)(ov * ftanh(cv));
    }
}

// ---- head GEMM: out = h @ W_head^T + b_head --------------------------------
__global__ void __launch_bounds__(128)
k_head(const __bf16* __restrict__ h_bf,     // [B, H]
       const __bf16* __restrict__ Whead,    // [D, H]
       const float*  __restrict__ b_head,   // [D]
       float* __restrict__ out) {           // [B, D]
    const int lane = threadIdx.x & 31;
    const int wave = threadIdx.x >> 5;
    const int m0 = wave * 16;
    const int n0 = blockIdx.x * 16;

    v8f acc = v8f{};
    for (int k0 = 0; k0 < H_; k0 += 32) {
        v16bf a = load_frag_A(h_bf, H_, m0, k0, lane);
        int nn = n0 + (lane & 15);
        int hi = lane >> 4;
        v16bf b = *(const v16bf*)(Whead + (size_t)nn * H_ + (size_t)(k0 + hi * 16));
        acc = wmma_bf16(a, b, acc);
    }
    const int hi = lane >> 4;
    const int n  = n0 + (lane & 15);
    const float bb = b_head[n];
#pragma unroll
    for (int r = 0; r < 8; ++r) {
        const int m = m0 + r + 8 * hi;
        out[(size_t)m * D_ + n] = acc[r] + bb;
    }
}

// ---- launch ----------------------------------------------------------------
extern "C" void kernel_launch(void* const* d_in, const int* in_sizes, int n_in,
                              void* d_out, int out_size, void* d_ws, size_t ws_size,
                              hipStream_t stream) {
    (void)in_sizes; (void)n_in; (void)out_size; (void)ws_size;
    const float* x      = (const float*)d_in[0];
    const float* W_ih   = (const float*)d_in[1];
    const float* W_hh   = (const float*)d_in[2];
    const float* b_ih   = (const float*)d_in[3];
    const float* b_hh   = (const float*)d_in[4];
    const float* W_head = (const float*)d_in[5];
    const float* b_head = (const float*)d_in[6];
    float* out = (float*)d_out;

    // workspace layout (bytes)
    char* ws = (char*)d_ws;
    const size_t szX     = (size_t)B_ * T_ * D_ * 2;   // 64 MB
    const size_t szWih   = (size_t)G4H_ * D_ * 2;      // 16 MB
    const size_t szWhh   = (size_t)G4H_ * H_ * 2;      // 32 MB
    const size_t szWhead = (size_t)D_ * H_ * 2;        //  4 MB
    const size_t szBias  = (size_t)G4H_ * 4;
    const size_t szHbuf  = (size_t)B_ * H_ * 2;        // bf16
    const size_t szCbuf  = (size_t)B_ * H_ * 4;        // f32

    __bf16* x_bf   = (__bf16*)(ws);
    __bf16* Wih_bf = (__bf16*)(ws + szX);
    __bf16* Whh_bf = (__bf16*)(ws + szX + szWih);
    __bf16* Whd_bf = (__bf16*)(ws + szX + szWih + szWhh);
    float*  bias   = (float*)(ws + szX + szWih + szWhh + szWhead);
    char*   state  = ws + szX + szWih + szWhh + szWhead + szBias;
    __bf16* hbuf[2] = {(__bf16*)state, (__bf16*)(state + szHbuf)};
    float*  cbuf[2] = {(float*)(state + 2 * szHbuf),
                       (float*)(state + 2 * szHbuf + szCbuf)};

    // convert inputs to bf16
    {
        long n;
        n = (long)B_ * T_ * D_;
        k_cvt_bf16<<<(unsigned)((n / 4 + 255) / 256), 256, 0, stream>>>(x, x_bf, n);
        n = (long)G4H_ * D_;
        k_cvt_bf16<<<(unsigned)((n / 4 + 255) / 256), 256, 0, stream>>>(W_ih, Wih_bf, n);
        n = (long)G4H_ * H_;
        k_cvt_bf16<<<(unsigned)((n / 4 + 255) / 256), 256, 0, stream>>>(W_hh, Whh_bf, n);
        n = (long)D_ * H_;
        k_cvt_bf16<<<(unsigned)((n / 4 + 255) / 256), 256, 0, stream>>>(W_head, Whd_bf, n);
    }
    k_bias_sum<<<(G4H_ + 255) / 256, 256, 0, stream>>>(b_ih, b_hh, bias, G4H_);

    // zero h/c state (both ping-pong buffers): 2*szHbuf + 2*szCbuf bytes
    {
        long nwords = (long)(2 * szHbuf + 2 * szCbuf) / 4;
        k_zero_u32<<<(unsigned)((nwords + 255) / 256), 256, 0, stream>>>(
            (unsigned int*)state, nwords);
    }

    // sequential recurrence: ping-pong h/c across 512 step kernels
    for (int t = 0; t < T_; ++t) {
        const int src = t & 1, dst = (t & 1) ^ 1;
        k_lstm_step<<<H_ / 16, 128, 0, stream>>>(
            x_bf, Wih_bf, Whh_bf, bias,
            hbuf[src], cbuf[src], hbuf[dst], cbuf[dst], t);
    }

    // head: final h lives in buffer (T_ & 1) after the last swap
    k_head<<<D_ / 16, 128, 0, stream>>>(hbuf[T_ & 1], Whd_bf, b_head, out);
}
